// PointcloudEncoder_46926812676419
// MI455X (gfx1250) — matrix operand order repaired and checked by reading, not compiled, not run on hardware
//
#include <hip/hip_runtime.h>

// ---------------------------------------------------------------------------
// Point-BERT style pipeline for MI455X (gfx1250, wave32, WMMA bf16).
// All matmuls go through v_wmma_f32_16x16x32_bf16 with f32 accumulation.
// A-tiles staged to LDS via gfx1250 async global->LDS copies when available.
// ---------------------------------------------------------------------------

#define B_    8
#define N_    8192
#define G_    512
#define K_    32
#define ENC_  512
#define TD_   768
#define ED_   1024
#define S_    513            // 1 cls + 512 tokens
#define M4_   4104           // B_*S_
#define HEADS_ 12
#define DH_   64
#define ATTP_ 520            // att row pitch (bf16, multiple of 8 -> 16B rows)
#define EPS_  1e-5f
#define OUT_PF_ELEMS 4194304 // 8*512*1024

typedef __attribute__((ext_vector_type(16))) __bf16 v16bf;
typedef __attribute__((ext_vector_type(8)))  float  v8f;
typedef int async_b128_t __attribute__((vector_size(16)));   // matches builtin param

#if defined(__HIP_DEVICE_COMPILE__) && __has_builtin(__builtin_amdgcn_global_load_async_to_lds_b128)
#define ASYNC_LDS_COPY 1
#else
#define ASYNC_LDS_COPY 0
#endif

__device__ __forceinline__ void wait_async0() {
#if defined(__HIP_DEVICE_COMPILE__)
#if __has_builtin(__builtin_amdgcn_s_wait_asynccnt)
  __builtin_amdgcn_s_wait_asynccnt(0);
#else
  asm volatile("s_wait_asynccnt 0x0" ::: "memory");
#endif
#endif
}

// ---------------------------------------------------------------------------
// WMMA helpers (A/B use contiguous k = 16*(lane/16) + s, wave32)
// ---------------------------------------------------------------------------
__device__ __forceinline__ v8f wmma_bf16(v16bf a, v16bf b, v8f c) {
  return __builtin_amdgcn_wmma_f32_16x16x32_bf16(false, a, false, b, (short)0, c,
                                                 false, false);
}

// A fragment: 16x32 bf16 tile, row-major source with pitch `lda` (bf16 elems).
__device__ __forceinline__ v16bf load_a_frag(const __bf16* A, int lda) {
  int t = threadIdx.x & 31;
  const __bf16* row = A + (t & 15) * lda;
  int kb = (t >> 4) * 16;
  v16bf a;
#pragma unroll
  for (int j = 0; j < 8; ++j) a[j] = row[kb + j];
#pragma unroll
  for (int j = 0; j < 8; ++j) a[8 + j] = row[kb + 8 + j];
  return a;
}

// B fragment (fast, fully in-bounds): 32x16 tile of B[k,n], 32-bit offsets so
// the backend can use SADDR-base + VGPR-offset addressing.
template <bool SRC_BF16>
__device__ __forceinline__ v16bf load_b_frag_fast(const void* Bv, int sBk, int sBn,
                                                  int k0, int n0) {
  int t = threadIdx.x & 31;
  int off = (k0 + ((t >> 4) << 4)) * sBk + (n0 + (t & 15)) * sBn;
  v16bf b;
#pragma unroll
  for (int j = 0; j < 16; ++j) {
    float v;
    if (SRC_BF16) v = (float)((const __bf16*)Bv)[off];
    else          v = ((const float*)Bv)[off];
    b[j] = (__bf16)v;
    off += sBk;
  }
  return b;
}

// B fragment (edge): branchless clamped addresses + cndmask-to-zero, 32-bit.
template <bool SRC_BF16>
__device__ __forceinline__ v16bf load_b_frag_guard(const void* Bv, int sBk, int sBn,
                                                   int k0, int n0, int Klim, int Nlim) {
  int t = threadIdx.x & 31;
  int n = n0 + (t & 15);
  int kb = k0 + ((t >> 4) << 4);
  bool nok = (n < Nlim);
  int nb = (nok ? n : (Nlim - 1)) * sBn;
  v16bf b;
#pragma unroll
  for (int j = 0; j < 16; ++j) {
    int k = kb + j;
    bool kok = (k < Klim);
    int kc = kok ? k : (Klim - 1);
    int off = kc * sBk + nb;
    float v;
    if (SRC_BF16) v = (float)((const __bf16*)Bv)[off];
    else          v = ((const float*)Bv)[off];
    if (!(kok && nok)) v = 0.f;
    b[j] = (__bf16)v;
  }
  return b;
}

__device__ __forceinline__ float gelu_exact(float x) {
  return 0.5f * x * (1.0f + erff(x * 0.70710678118654752f));
}

// ---------------------------------------------------------------------------
// Generic batched WMMA GEMM:  C[m,n] = act(alpha * sum_k A[m,k]*B[k,n] + bias[n])
// A: bf16, addr = A + m*lda + k; B: f32/bf16, addr = B + k*sBk + n*sBn; C: f32.
// grid: (ceil(N/64), ceil(M/128), BB*BH), block 256 = 8 waves, wave = 16 rows.
// Per-matrix offsets fit in 32 bits (largest tensor ~12.6M elems).
// ---------------------------------------------------------------------------
__global__ void __launch_bounds__(256)
gemm_wmma_kernel(const __bf16* A, long lda, long sAb, long sAh,
                 const void* B, int bIsF32, long sBk, long sBn, long sBb, long sBh,
                 float* C, long ldc, long sCb, long sCh,
                 const float* bias, int act, float alpha,
                 int M, int N, int K, int BH) {
  __shared__ __bf16 sA[128 * 40];     // 128 rows x 32 k, pitch 40

  int bz = blockIdx.z;
  int bb = bz / BH, bh = bz % BH;
  const __bf16* Ab = A + (long)bb * sAb + (long)bh * sAh;
  const void* Bb;
  if (bIsF32) Bb = (const void*)((const float*)B + (long)bb * sBb + (long)bh * sBh);
  else        Bb = (const void*)((const __bf16*)B + (long)bb * sBb + (long)bh * sBh);
  float* Cb = C + (long)bb * sCb + (long)bh * sCh;

  const int ldai = (int)lda, ldci = (int)ldc, sBki = (int)sBk, sBni = (int)sBn;

  int m0 = blockIdx.y * 128;
  int n0 = blockIdx.x * 64;
  int wave = threadIdx.x >> 5;
  int lane = threadIdx.x & 31;
  int r  = threadIdx.x >> 1;           // staging row 0..127
  int c0 = (threadIdx.x & 1) * 16;     // staging col block {0,16}

  const bool rowAligned = ((ldai & 7) == 0);   // 16B-aligned bf16 rows
  v8f a0 = {}, a1 = {}, a2 = {}, a3 = {};

  for (int k0 = 0; k0 < K; k0 += 32) {
    // ---- stage A tile (128x32 bf16) into LDS
    bool fullA = (m0 + 128 <= M) && (k0 + 32 <= K) && rowAligned;  // uniform
    if (fullA) {
      const __bf16* gsrc = Ab + (m0 + r) * ldai + (k0 + c0);
      __bf16* ldst = &sA[r * 40 + c0];
#if ASYNC_LDS_COPY
      __builtin_amdgcn_global_load_async_to_lds_b128(
          (async_b128_t*)gsrc, (async_b128_t*)ldst, 0, 0);
      __builtin_amdgcn_global_load_async_to_lds_b128(
          (async_b128_t*)(gsrc + 8), (async_b128_t*)(ldst + 8), 0, 0);
      wait_async0();
#else
#pragma unroll
      for (int j = 0; j < 16; ++j) ldst[j] = gsrc[j];
#endif
    } else {
      int m  = m0 + r;
      bool mok = (m < M);
      int mc = mok ? m : (M - 1);
      int rowOff = mc * ldai;
#pragma unroll
      for (int j = 0; j < 16; ++j) {
        int k = k0 + c0 + j;
        bool kok = (k < K);
        int kc = kok ? k : (K - 1);
        __bf16 v = Ab[rowOff + kc];               // unconditional clamped load
        if (!(mok && kok)) v = (__bf16)0.f;       // cndmask, no exec churn
        sA[r * 40 + c0 + j] = v;
      }
    }
    __syncthreads();

#if defined(__gfx1250__)
    if (k0 + 32 < K) {   // hint the next B k-slab toward L2 (global_prefetch)
      long esz = bIsF32 ? 4 : 2;
      __builtin_prefetch((const char*)Bb + ((long)(k0 + 32) * sBki + (long)n0 * sBni) * esz, 0, 0);
    }
#endif

    v16bf a = load_a_frag(&sA[wave * 16 * 40], 40);
    bool fullB = (n0 + 64 <= N) && (k0 + 32 <= K);   // uniform
    if (fullB) {
      if (bIsF32) {
        a0 = wmma_bf16(a, load_b_frag_fast<false>(Bb, sBki, sBni, k0, n0     ), a0);
        a1 = wmma_bf16(a, load_b_frag_fast<false>(Bb, sBki, sBni, k0, n0 + 16), a1);
        a2 = wmma_bf16(a, load_b_frag_fast<false>(Bb, sBki, sBni, k0, n0 + 32), a2);
        a3 = wmma_bf16(a, load_b_frag_fast<false>(Bb, sBki, sBni, k0, n0 + 48), a3);
      } else {
        a0 = wmma_bf16(a, load_b_frag_fast<true >(Bb, sBki, sBni, k0, n0     ), a0);
        a1 = wmma_bf16(a, load_b_frag_fast<true >(Bb, sBki, sBni, k0, n0 + 16), a1);
        a2 = wmma_bf16(a, load_b_frag_fast<true >(Bb, sBki, sBni, k0, n0 + 32), a2);
        a3 = wmma_bf16(a, load_b_frag_fast<true >(Bb, sBki, sBni, k0, n0 + 48), a3);
      }
    } else {
      if (bIsF32) {
        a0 = wmma_bf16(a, load_b_frag_guard<false>(Bb, sBki, sBni, k0, n0     , K, N), a0);
        a1 = wmma_bf16(a, load_b_frag_guard<false>(Bb, sBki, sBni, k0, n0 + 16, K, N), a1);
        a2 = wmma_bf16(a, load_b_frag_guard<false>(Bb, sBki, sBni, k0, n0 + 32, K, N), a2);
        a3 = wmma_bf16(a, load_b_frag_guard<false>(Bb, sBki, sBni, k0, n0 + 48, K, N), a3);
      } else {
        a0 = wmma_bf16(a, load_b_frag_guard<true >(Bb, sBki, sBni, k0, n0     , K, N), a0);
        a1 = wmma_bf16(a, load_b_frag_guard<true >(Bb, sBki, sBni, k0, n0 + 16, K, N), a1);
        a2 = wmma_bf16(a, load_b_frag_guard<true >(Bb, sBki, sBni, k0, n0 + 32, K, N), a2);
        a3 = wmma_bf16(a, load_b_frag_guard<true >(Bb, sBki, sBni, k0, n0 + 48, K, N), a3);
      }
    }
    __syncthreads();
  }

  // epilogue (C layout: vgpr r, lanes 0-15 -> M=r, lanes 16-31 -> M=8+r, N=lane&15)
  int mo = m0 + wave * 16 + ((lane >> 4) << 3);
  v8f accs[4] = {a0, a1, a2, a3};
#pragma unroll
  for (int nt = 0; nt < 4; ++nt) {
    int n = n0 + nt * 16 + (lane & 15);
#pragma unroll
    for (int rr = 0; rr < 8; ++rr) {
      int m = mo + rr;
      if (m < M && n < N) {
        float v = accs[nt][rr] * alpha;
        if (bias) v += bias[n];
        if (act == 1) v = fmaxf(v, 0.f);
        else if (act == 2) v = gelu_exact(v);
        Cb[m * ldci + n] = v;
      }
    }
  }
}

// ---------------------------------------------------------------------------
// FPS: one 1024-thread block per batch; distances register-resident (8/thread);
// 512 dependent steps of min-update + block argmax (tie -> lower index).
// ---------------------------------------------------------------------------
__global__ void __launch_bounds__(1024)
fps_kernel(const float* pts, int* fidx) {
  int b = blockIdx.x;
  const float* P = pts + (long)b * N_ * 3;
  __shared__ float smax[1024];
  __shared__ int   sidx[1024];
  __shared__ float scen[3];
  int t = threadIdx.x;
  float dist[8];
#pragma unroll
  for (int j = 0; j < 8; ++j) dist[j] = 1e10f;
  int far = 0;
  for (int it = 0; it < G_; ++it) {
    if (t == 0) {
      fidx[b * G_ + it] = far;
      scen[0] = P[far * 3 + 0]; scen[1] = P[far * 3 + 1]; scen[2] = P[far * 3 + 2];
    }
    __syncthreads();
    float cx = scen[0], cy = scen[1], cz = scen[2];
    float bmax = -1.f; int bidx = 0;
#pragma unroll
    for (int j = 0; j < 8; ++j) {
      int i = j * 1024 + t;
      float dx = P[i * 3 + 0] - cx, dy = P[i * 3 + 1] - cy, dz = P[i * 3 + 2] - cz;
      float d = dx * dx + dy * dy + dz * dz;
      dist[j] = fminf(dist[j], d);
      if (dist[j] > bmax) { bmax = dist[j]; bidx = i; }
    }
    smax[t] = bmax; sidx[t] = bidx;
    __syncthreads();
    for (int s = 512; s > 0; s >>= 1) {
      if (t < s) {
        if (smax[t + s] > smax[t] ||
            (smax[t + s] == smax[t] && sidx[t + s] < sidx[t])) {
          smax[t] = smax[t + s]; sidx[t] = sidx[t + s];
        }
      }
      __syncthreads();
    }
    far = sidx[0];   // next iteration's first barrier orders the re-writes
  }
}

__global__ void __launch_bounds__(256)
centers_kernel(const float* pts, const int* fidx, float* cF, __bf16* cH) {
  int i = blockIdx.x * 256 + threadIdx.x;
  if (i >= B_ * G_) return;
  int b = i >> 9;
  int fi = fidx[i];
#pragma unroll
  for (int c = 0; c < 3; ++c) {
    float v = pts[((long)b * N_ + fi) * 3 + c];
    cF[i * 3 + c] = v;
    cH[i * 3 + c] = (__bf16)v;
  }
}

// ---------------------------------------------------------------------------
// KNN top-32 (ascending distance, tie -> lower index): distances in LDS (32KB),
// 32 rounds of masked argmin. One 256-thread block per (b,g).
// ---------------------------------------------------------------------------
__global__ void __launch_bounds__(256)
knn_kernel(const float* pts, const float* centers, int* knn_out) {
  int bg = blockIdx.x;
  int b = bg >> 9;
  __shared__ float sd[N_];
  __shared__ float smin[256];
  __shared__ int   sidx[256];
  int t = threadIdx.x;
  float cx = centers[bg * 3 + 0], cy = centers[bg * 3 + 1], cz = centers[bg * 3 + 2];
  float c2 = cx * cx + cy * cy + cz * cz;
  for (int i = t; i < N_; i += 256) {
    float px = pts[((long)b * N_ + i) * 3 + 0];
    float py = pts[((long)b * N_ + i) * 3 + 1];
    float pz = pts[((long)b * N_ + i) * 3 + 2];
    float p2 = px * px + py * py + pz * pz;
    sd[i] = c2 + p2 - 2.f * (cx * px + cy * py + cz * pz);  // matches reference
  }
  __syncthreads();
  for (int kk = 0; kk < K_; ++kk) {
    float bm = 3.4e38f; int bi = 0;
    for (int i = t; i < N_; i += 256) {
      float d = sd[i];
      if (d < bm) { bm = d; bi = i; }
    }
    smin[t] = bm; sidx[t] = bi;
    __syncthreads();
    for (int s = 128; s > 0; s >>= 1) {
      if (t < s) {
        if (smin[t + s] < smin[t] ||
            (smin[t + s] == smin[t] && sidx[t + s] < sidx[t])) {
          smin[t] = smin[t + s]; sidx[t] = sidx[t + s];
        }
      }
      __syncthreads();
    }
    if (t == 0) {
      knn_out[bg * K_ + kk] = sidx[0];
      sd[sidx[0]] = 3.4e38f;
    }
    __syncthreads();
  }
}

// ---------------------------------------------------------------------------
// Fused per-group PointNet encoder. One 128-thread block (4 waves) per (b,g).
// All intermediates double-buffered in 64KB LDS; every matmul on WMMA.
// ---------------------------------------------------------------------------
__device__ __forceinline__ void store_tile_bn_relu(__bf16* dst, int ldd, int mrow, int n0,
                                                   v8f acc, const float* bias,
                                                   const float* g, const float* bb,
                                                   const float* mm, const float* vv) {
  int lane = threadIdx.x & 31;
  int n = n0 + (lane & 15);
  int mo = mrow + ((lane >> 4) << 3);
  float bi = bias[n], gg = g[n], b2 = bb[n], mu = mm[n];
  float iv = rsqrtf(vv[n] + EPS_);
#pragma unroll
  for (int r = 0; r < 8; ++r) {
    float v = (acc[r] + bi - mu) * iv * gg + b2;
    dst[(mo + r) * ldd + n] = (__bf16)fmaxf(v, 0.f);
  }
}

__device__ __forceinline__ void store_tile_bias_f32(float* dst, int ldd, int mrow, int n0,
                                                    v8f acc, const float* bias) {
  int lane = threadIdx.x & 31;
  int n = n0 + (lane & 15);
  int mo = mrow + ((lane >> 4) << 3);
  float bi = bias[n];
#pragma unroll
  for (int r = 0; r < 8; ++r) dst[(mo + r) * ldd + n] = acc[r] + bi;
}

__global__ void __launch_bounds__(128)
encoder_kernel(const float* pts, const float* colors, const int* knn, const float* centers,
               const float* w1, const float* b1,
               const float* g1, const float* bb1, const float* m1, const float* v1,
               const float* w2, const float* b2,
               const float* w3, const float* b3,
               const float* g2, const float* bb2, const float* m2, const float* v2,
               const float* w4, const float* b4,
               __bf16* tokens) {
  __shared__ float  bufA[32 * 256];    // 32KB: feat(bf16) -> h2(f32) -> h3(bf16)
  __shared__ __bf16 bufB[32 * 512];    // 32KB: h1(bf16)   -> cat(bf16)
  __bf16* sFeat = (__bf16*)bufA;       // 32x32
  __bf16* sH1   = bufB;                // 32x128
  float*  sH2   = bufA;                // 32x256
  __bf16* sCat  = bufB;                // 32x512
  __bf16* sH3   = (__bf16*)bufA;       // 32x512

  int bg = blockIdx.x;
  int b = bg >> 9;
  int wave = threadIdx.x >> 5;
  int lane = threadIdx.x & 31;

  // ---- gather features: [nb - center, colors], pad K dim 6 -> 32 with zeros
  for (int i = threadIdx.x; i < 32 * 32; i += 128) sFeat[i] = (__bf16)0.f;
  __syncthreads();
  if (threadIdx.x < K_) {
    int i = threadIdx.x;
    int pi = knn[bg * K_ + i];
#pragma unroll
    for (int c = 0; c < 3; ++c) {
      float pv = pts[((long)b * N_ + pi) * 3 + c] - centers[bg * 3 + c];
      float cv = colors[((long)b * N_ + pi) * 3 + c];
      sFeat[i * 32 + c]     = (__bf16)pv;
      sFeat[i * 32 + 3 + c] = (__bf16)cv;
    }
  }
  __syncthreads();

  // ---- L1: 6->128, bn1+relu  (tiles: 2 m x 8 n; wave does 2 n-tiles)
  for (int ntl = wave * 2; ntl < wave * 2 + 2; ++ntl) {
#pragma unroll
    for (int mt = 0; mt < 2; ++mt) {
      v8f acc = {};
      v16bf a = load_a_frag(sFeat + mt * 16 * 32, 32);
      v16bf bf = load_b_frag_guard<false>(w1, 128, 1, 0, ntl * 16, 6, 128);
      acc = wmma_bf16(a, bf, acc);
      store_tile_bn_relu(sH1, 128, mt * 16, ntl * 16, acc, b1, g1, bb1, m1, v1);
    }
  }
  __syncthreads();

  // ---- L2: 128->256 + bias (f32 to LDS)  (2 m x 16 n; wave does 4 n-tiles)
  for (int ntl = wave * 4; ntl < wave * 4 + 4; ++ntl) {
#pragma unroll
    for (int mt = 0; mt < 2; ++mt) {
      v8f acc = {};
      for (int k0 = 0; k0 < 128; k0 += 32) {
        v16bf a = load_a_frag(sH1 + mt * 16 * 128 + k0, 128);
        v16bf bf = load_b_frag_fast<false>(w2, 256, 1, k0, ntl * 16);
        acc = wmma_bf16(a, bf, acc);
      }
      store_tile_bias_f32(sH2, 256, mt * 16, ntl * 16, acc, b2);
    }
  }
  __syncthreads();

  // ---- gmax over 32 points + concat [gmax, h2] -> 32x512 bf16
  for (int c = threadIdx.x; c < 256; c += 128) {
    float mx = -3.4e38f;
    for (int m = 0; m < 32; ++m) mx = fmaxf(mx, sH2[m * 256 + c]);
    __bf16 mb = (__bf16)mx;
    for (int m = 0; m < 32; ++m) {
      sCat[m * 512 + c]       = mb;
      sCat[m * 512 + 256 + c] = (__bf16)sH2[m * 256 + c];
    }
  }
  __syncthreads();

  // ---- L3: 512->512, bn2+relu  (wave: 8 n-tiles, acc-blocked over n)
  for (int mt = 0; mt < 2; ++mt) {
    v8f acc[8] = {};
    for (int k0 = 0; k0 < 512; k0 += 32) {
      v16bf a = load_a_frag(sCat + mt * 16 * 512 + k0, 512);
#pragma unroll
      for (int nt = 0; nt < 8; ++nt) {
        v16bf bf = load_b_frag_fast<false>(w3, 512, 1, k0, (wave * 8 + nt) * 16);
        acc[nt] = wmma_bf16(a, bf, acc[nt]);
      }
    }
#pragma unroll
    for (int nt = 0; nt < 8; ++nt)
      store_tile_bn_relu(sH3, 512, mt * 16, (wave * 8 + nt) * 16, acc[nt],
                         b3, g2, bb2, m2, v2);
    __syncthreads();
  }

  // ---- L4: 512->512 + bias, max over 32 points -> token
  float cm[8];
#pragma unroll
  for (int nt = 0; nt < 8; ++nt) cm[nt] = -3.4e38f;
  for (int mt = 0; mt < 2; ++mt) {
    v8f acc[8] = {};
    for (int k0 = 0; k0 < 512; k0 += 32) {
      v16bf a = load_a_frag(sH3 + mt * 16 * 512 + k0, 512);
#pragma unroll
      for (int nt = 0; nt < 8; ++nt) {
        v16bf bf = load_b_frag_fast<false>(w4, 512, 1, k0, (wave * 8 + nt) * 16);
        acc[nt] = wmma_bf16(a, bf, acc[nt]);
      }
    }
#pragma unroll
    for (int nt = 0; nt < 8; ++nt) {
      float m = -3.4e38f;
#pragma unroll
      for (int r = 0; r < 8; ++r) m = fmaxf(m, acc[nt][r]);
      cm[nt] = fmaxf(cm[nt], m);
    }
  }
#pragma unroll
  for (int nt = 0; nt < 8; ++nt) {
    float m = fmaxf(cm[nt], __shfl_xor(cm[nt], 16, 32));
    int n = (wave * 8 + nt) * 16 + (lane & 15);
    if (lane < 16) tokens[(long)bg * ENC_ + n] = (__bf16)(m + b4[n]);
  }
}

// ---------------------------------------------------------------------------
// Elementwise / normalization kernels
// ---------------------------------------------------------------------------
__global__ void __launch_bounds__(256)
f32_to_bf16_kernel(__bf16* d, const float* s, long n) {
  long i = (long)blockIdx.x * 256 + threadIdx.x;
  long st = (long)gridDim.x * 256;
  for (; i < n; i += st) d[i] = (__bf16)s[i];
}

__global__ void __launch_bounds__(256)
add_inplace_kernel(float* x, const float* t, long n) {
  long i = (long)blockIdx.x * 256 + threadIdx.x;
  long st = (long)gridDim.x * 256;
  for (; i < n; i += st) x[i] += t[i];
}

__global__ void __launch_bounds__(256)
copy_f32_kernel(float* d, const float* s, long n) {
  long i = (long)blockIdx.x * 256 + threadIdx.x;
  long st = (long)gridDim.x * 256;
  for (; i < n; i += st) d[i] = s[i];
}

__global__ void __launch_bounds__(256)
build_x_kernel(const float* tokE, const float* posE,
               const float* clsTok, const float* clsPos,
               float* x, float* x0) {
  long i = (long)blockIdx.x * 256 + threadIdx.x;
  long n = (long)M4_ * TD_;
  long st = (long)gridDim.x * 256;
  for (; i < n; i += st) {
    int r = (int)(i / TD_), d = (int)(i % TD_);
    int b = r / S_, s = r % S_;
    if (s == 0) {
      x[i] = clsTok[d] + clsPos[d];
    } else {
      long tr = (long)(b * G_ + (s - 1)) * TD_ + d;
      float v = tokE[tr] + posE[tr];
      x[i] = v;
      x0[tr] = v;
    }
  }
}

// one wave per row layernorm; optional f32 and/or bf16 outputs
__global__ void __launch_bounds__(256)
layernorm_kernel(const float* in, const float* g, const float* b,
                 int rows, int D, float* outF, __bf16* outB) {
  int row = blockIdx.x * 8 + (threadIdx.x >> 5);
  int lane = threadIdx.x & 31;
  if (row >= rows) return;
  const float* p = in + (long)row * D;
  float s = 0.f, s2 = 0.f;
  for (int c = lane; c < D; c += 32) { float v = p[c]; s += v; s2 += v * v; }
#pragma unroll
  for (int m = 16; m; m >>= 1) { s += __shfl_xor(s, m, 32); s2 += __shfl_xor(s2, m, 32); }
  float mu = s / D;
  float var = s2 / D - mu * mu;
  float rs = rsqrtf(var + EPS_);
  for (int c = lane; c < D; c += 32) {
    float y = (p[c] - mu) * rs * g[c] + b[c];
    if (outF) outF[(long)row * D + c] = y;
    if (outB) outB[(long)row * D + c] = (__bf16)y;
  }
}

// one wave per row softmax, writes bf16 (padded pitch) for the att@V WMMA
__global__ void __launch_bounds__(256)
softmax_kernel(const float* S, __bf16* A, int rows, int cols, int opitch) {
  int row = blockIdx.x * 8 + (threadIdx.x >> 5);
  int lane = threadIdx.x & 31;
  if (row >= rows) return;
  const float* p = S + (long)row * cols;
  float m = -3.4e38f;
  for (int c = lane; c < cols; c += 32) m = fmaxf(m, p[c]);
#pragma unroll
  for (int s = 16; s; s >>= 1) m = fmaxf(m, __shfl_xor(m, s, 32));
  float sum = 0.f;
  for (int c = lane; c < cols; c += 32) sum += expf(p[c] - m);
#pragma unroll
  for (int s = 16; s; s >>= 1) sum += __shfl_xor(sum, s, 32);
  float inv = 1.f / sum;
  __bf16* q = A + (long)row * opitch;
  for (int c = lane; c < cols; c += 32) q[c] = (__bf16)(expf(p[c] - m) * inv);
}

// pf = x0 + ln(x4) + ln(x8) + 100*ln(x12) + 20*ln(x12)[cls], then fcn layernorm
__global__ void __launch_bounds__(256)
combine_kernel(const float* x0, const float* x4n, const float* x8n, const float* x12n,
               const float* fg, const float* fb, __bf16* out) {
  __shared__ float pf[TD_];
  __shared__ float red[256];
  __shared__ float stats[2];
  int blk = blockIdx.x;            // b*512 + g
  int b = blk >> 9;
  int t = threadIdx.x;
  long r  = (long)(b * S_ + 1 + (blk & 511)) * TD_;
  long rc = (long)(b * S_) * TD_;
  float s = 0.f, s2 = 0.f;
  for (int d = t; d < TD_; d += 256) {
    float v = x0[(long)blk * TD_ + d] + x4n[r + d] + x8n[r + d]
              + 100.f * x12n[r + d] + 20.f * x12n[rc + d];
    pf[d] = v; s += v; s2 += v * v;
  }
  red[t] = s; __syncthreads();
  for (int st = 128; st; st >>= 1) { if (t < st) red[t] += red[t + st]; __syncthreads(); }
  if (t == 0) stats[0] = red[0];
  __syncthreads();
  red[t] = s2; __syncthreads();
  for (int st = 128; st; st >>= 1) { if (t < st) red[t] += red[t + st]; __syncthreads(); }
  if (t == 0) stats[1] = red[0];
  __syncthreads();
  float mu = stats[0] / TD_;
  float var = stats[1] / TD_ - mu * mu;
  float rs = rsqrtf(var + EPS_);
  for (int d = t; d < TD_; d += 256)
    out[(long)blk * TD_ + d] = (__bf16)((pf[d] - mu) * rs * fg[d] + fb[d]);
}

// ---------------------------------------------------------------------------
// Host orchestration
// ---------------------------------------------------------------------------
extern "C" void kernel_launch(void* const* d_in, const int* in_sizes, int n_in,
                              void* d_out, int out_size, void* d_ws, size_t ws_size,
                              hipStream_t stream) {
  const float* pts    = (const float*)d_in[0];
  const float* colors = (const float*)d_in[1];
  auto W = [&](int i) { return (const float*)d_in[i]; };
  // param index map (setup_inputs insertion order):
  // 2:e_w1 3:e_b1 4..7:bn1(g,b,m,v) 8:e_w2 9:e_b2 10:e_w3 11:e_b3 12..15:bn2
  // 16:e_w4 17:e_b4 18:e2t_w 19:e2t_b 20:t2e_w 21:t2e_b 22:cls_token 23:cls_pos
  // 24:pe_w1 25:pe_b1 26:pe_w2 27:pe_b2  blocks: 28 + 12*i  (ln1g,ln1b,qkv_w,
  // qkv_b,proj_w,proj_b,ln2g,ln2b,fc1_w,fc1_b,fc2_w,fc2_b)  172..175: norm/fcn

  // --- workspace arena
  char* base = (char*)d_ws;
  size_t off = 0;
  auto alloc = [&](size_t bytes) -> char* {
    off = (off + 255) & ~(size_t)255;
    char* p = base + off; off += bytes; return p;
  };
  int*    fidx     = (int*)alloc((size_t)B_ * G_ * 4);
  float*  centersF = (float*)alloc((size_t)B_ * G_ * 3 * 4);
  __bf16* centersH = (__bf16*)alloc((size_t)B_ * G_ * 3 * 2);
  __bf16* tokens   = (__bf16*)alloc((size_t)B_ * G_ * ENC_ * 2);
  float*  tokE     = (float*)alloc((size_t)B_ * G_ * TD_ * 4);
  float*  peh      = (float*)alloc((size_t)B_ * G_ * 128 * 4);
  __bf16* pehB     = (__bf16*)alloc((size_t)B_ * G_ * 128 * 2);
  float*  posE     = (float*)alloc((size_t)B_ * G_ * TD_ * 4);
  float*  x        = (float*)alloc((size_t)M4_ * TD_ * 4);
  float*  x0       = (float*)alloc((size_t)B_ * G_ * TD_ * 4);
  float*  x4       = (float*)alloc((size_t)M4_ * TD_ * 4);
  float*  x8       = (float*)alloc((size_t)M4_ * TD_ * 4);
  __bf16* xnB      = (__bf16*)alloc((size_t)M4_ * TD_ * 2);
  float*  qkvF     = (float*)alloc((size_t)M4_ * 3 * TD_ * 4);
  __bf16* qkvB     = (__bf16*)alloc((size_t)M4_ * 3 * TD_ * 2);
  float*  scoresF  = (float*)alloc((size_t)B_ * HEADS_ * S_ * S_ * 4);
  __bf16* attB     = (__bf16*)alloc((size_t)B_ * HEADS_ * S_ * ATTP_ * 2);
  float*  obufF    = (float*)alloc((size_t)M4_ * TD_ * 4);
  __bf16* obufB    = (__bf16*)alloc((size_t)M4_ * TD_ * 2);
  float*  tmpF     = (float*)alloc((size_t)M4_ * TD_ * 4);
  float*  hF       = (float*)alloc((size_t)M4_ * 4 * TD_ * 4);
  __bf16* hB       = (__bf16*)alloc((size_t)M4_ * 4 * TD_ * 2);
  float*  x4n      = (float*)alloc((size_t)M4_ * TD_ * 4);
  float*  x8n      = (float*)alloc((size_t)M4_ * TD_ * 4);
  float*  x12n     = (float*)alloc((size_t)M4_ * TD_ * 4);
  __bf16* pfB      = (__bf16*)alloc((size_t)B_ * G_ * TD_ * 2);

  int* knnOut = ((int*)d_out) + OUT_PF_ELEMS;   // knn_idx output tail (bit-cast)

  auto gemm = [&](const __bf16* A, long lda, long sAb, long sAh,
                  const void* B, int bF32, long sBk, long sBn, long sBb, long sBh,
                  float* C, long ldc, long sCb, long sCh,
                  const float* bias, int act, float alpha,
                  int M, int N, int K, int BB, int BH) {
    dim3 g((N + 63) / 64, (M + 127) / 128, BB * BH);
    gemm_wmma_kernel<<<g, 256, 0, stream>>>(A, lda, sAb, sAh, B, bF32, sBk, sBn,
                                            sBb, sBh, C, ldc, sCb, sCh,
                                            bias, act, alpha, M, N, K, BH);
  };
  auto cvt = [&](__bf16* d, const float* s, long n) {
    f32_to_bf16_kernel<<<4096, 256, 0, stream>>>(d, s, n);
  };

  // ---- geometry stage
  fps_kernel<<<B_, 1024, 0, stream>>>(pts, fidx);
  centers_kernel<<<(B_ * G_ + 255) / 256, 256, 0, stream>>>(pts, fidx, centersF, centersH);
  knn_kernel<<<B_ * G_, 256, 0, stream>>>(pts, centersF, knnOut);

  // ---- fused group encoder -> tokens (bf16, 4096x512)
  encoder_kernel<<<B_ * G_, 128, 0, stream>>>(
      pts, colors, knnOut, centersF,
      W(2), W(3), W(4), W(5), W(6), W(7),
      W(8), W(9),
      W(10), W(11), W(12), W(13), W(14), W(15),
      W(16), W(17), tokens);

  // ---- token projection + positional embedding
  gemm(tokens, ENC_, 0, 0, W(18), 1, TD_, 1, 0, 0, tokE, TD_, 0, 0,
       W(19), 0, 1.f, B_ * G_, TD_, ENC_, 1, 1);
  gemm(centersH, 3, 0, 0, W(24), 1, 128, 1, 0, 0, peh, 128, 0, 0,
       W(25), 2, 1.f, B_ * G_, 128, 3, 1, 1);
  cvt(pehB, peh, (long)B_ * G_ * 128);
  gemm(pehB, 128, 0, 0, W(26), 1, TD_, 1, 0, 0, posE, TD_, 0, 0,
       W(27), 0, 1.f, B_ * G_, TD_, 128, 1, 1);

  build_x_kernel<<<4096, 256, 0, stream>>>(tokE, posE, W(22), W(23), x, x0);

  // ---- 12 transformer blocks
  const long nXT = (long)M4_ * TD_;
  for (int i = 0; i < 12; ++i) {
    int pb = 28 + 12 * i;
    layernorm_kernel<<<(M4_ + 7) / 8, 256, 0, stream>>>(x, W(pb), W(pb + 1),
                                                        M4_, TD_, nullptr, xnB);
    gemm(xnB, TD_, 0, 0, W(pb + 2), 1, 3 * TD_, 1, 0, 0, qkvF, 3 * TD_, 0, 0,
         W(pb + 3), 0, 1.f, M4_, 3 * TD_, TD_, 1, 1);
    cvt(qkvB, qkvF, (long)M4_ * 3 * TD_);

    // scores[b,h,q,k] = 0.125 * q . k     (batched over 8x12)
    gemm(qkvB, 3 * TD_, (long)S_ * 3 * TD_, DH_,
         qkvB + TD_, 0, 3 * TD_, 1, (long)S_ * 3 * TD_, DH_,
         scoresF, S_, (long)HEADS_ * S_ * S_, (long)S_ * S_,
         nullptr, 0, 0.125f, S_, S_, DH_, B_, HEADS_);
    softmax_kernel<<<(B_ * HEADS_ * S_ + 7) / 8, 256, 0, stream>>>(
        scoresF, attB, B_ * HEADS_ * S_, S_, ATTP_);

    // o[b,s,h*64+d] = att @ v
    gemm(attB, ATTP_, (long)HEADS_ * S_ * ATTP_, (long)S_ * ATTP_,
         qkvB + 2 * TD_, 0, 3 * TD_, 1, (long)S_ * 3 * TD_, DH_,
         obufF, TD_, (long)S_ * TD_, DH_,
         nullptr, 0, 1.f, S_, DH_, S_, B_, HEADS_);
    cvt(obufB, obufF, nXT);

    gemm(obufB, TD_, 0, 0, W(pb + 4), 1, TD_, 1, 0, 0, tmpF, TD_, 0, 0,
         W(pb + 5), 0, 1.f, M4_, TD_, TD_, 1, 1);
    add_inplace_kernel<<<4096, 256, 0, stream>>>(x, tmpF, nXT);

    layernorm_kernel<<<(M4_ + 7) / 8, 256, 0, stream>>>(x, W(pb + 6), W(pb + 7),
                                                        M4_, TD_, nullptr, xnB);
    gemm(xnB, TD_, 0, 0, W(pb + 8), 1, 4 * TD_, 1, 0, 0, hF, 4 * TD_, 0, 0,
         W(pb + 9), 2, 1.f, M4_, 4 * TD_, TD_, 1, 1);
    cvt(hB, hF, (long)M4_ * 4 * TD_);
    gemm(hB, 4 * TD_, 0, 0, W(pb + 10), 1, TD_, 1, 0, 0, tmpF, TD_, 0, 0,
         W(pb + 11), 0, 1.f, M4_, TD_, 4 * TD_, 1, 1);
    add_inplace_kernel<<<4096, 256, 0, stream>>>(x, tmpF, nXT);

    if (i == 3) copy_f32_kernel<<<4096, 256, 0, stream>>>(x4, x, nXT);
    if (i == 7) copy_f32_kernel<<<4096, 256, 0, stream>>>(x8, x, nXT);
  }

  // ---- feature pyramid combine + final projection
  layernorm_kernel<<<(M4_ + 7) / 8, 256, 0, stream>>>(x4, W(172), W(173), M4_, TD_, x4n, nullptr);
  layernorm_kernel<<<(M4_ + 7) / 8, 256, 0, stream>>>(x8, W(172), W(173), M4_, TD_, x8n, nullptr);
  layernorm_kernel<<<(M4_ + 7) / 8, 256, 0, stream>>>(x,  W(172), W(173), M4_, TD_, x12n, nullptr);
  combine_kernel<<<B_ * G_, 256, 0, stream>>>(x0, x4n, x8n, x12n, W(174), W(175), pfB);

  gemm(pfB, TD_, 0, 0, W(20), 1, ED_, 1, 0, 0, (float*)d_out, ED_, 0, 0,
       W(21), 0, 1.f, B_ * G_, ED_, TD_, 1, 1);
}
